// KuramotoInteraction_54511724921354
// MI455X (gfx1250) — matrix operand order repaired and checked by reading, not compiled, not run on hardware
//
#include <hip/hip_runtime.h>
#include <hip/hip_bf16.h>
#include <math.h>

// ---------------------------------------------------------------------------
// KuramotoInteraction on MI455X (gfx1250), wave32 + WMMA f16->f32.
// B=4, T=2048, D=512, H=8, hd=64.
// ---------------------------------------------------------------------------

typedef __attribute__((ext_vector_type(16))) _Float16 v16h;
typedef __attribute__((ext_vector_type(8)))  float    v8f;
typedef __attribute__((ext_vector_type(4)))  int      v4i;

#define BB   4
#define TT   2048
#define DD   512
#define HH   8
#define HD   64

// ------------------------- CDNA5 async-to-LDS path -------------------------
#if defined(__gfx1250__) && __has_builtin(__builtin_amdgcn_global_load_async_to_lds_b128)
#define ASYNC_LDS 1
typedef __attribute__((address_space(1))) v4i* gbl_v4p;   // global int4*
typedef __attribute__((address_space(3))) v4i* lds_v4p;   // LDS int4*
static __device__ __forceinline__ void async_copy_b128(void* lds, const void* g) {
  // emits global_load_async_to_lds_b128 (tracked by ASYNCcnt)
  __builtin_amdgcn_global_load_async_to_lds_b128((gbl_v4p)(v4i*)(void*)g,
                                                 (lds_v4p)(v4i*)lds, 0, 0);
}
#else
#define ASYNC_LDS 0
#endif

template <int N>
static __device__ __forceinline__ void wait_async_le() {
#if defined(__gfx1250__) && __has_builtin(__builtin_amdgcn_s_wait_asynccnt)
  __builtin_amdgcn_s_wait_asynccnt(N);
#else
  asm volatile("s_wait_asynccnt %0" :: "i"(N) : "memory");
#endif
}

static __device__ __forceinline__ v8f wmma_f16(v16h a, v16h b, v8f c) {
  return __builtin_amdgcn_wmma_f32_16x16x32_f16(
      /*neg_a=*/false, a, /*neg_b=*/false, b,
      /*c_mod=*/(short)0, c, /*reuse_a=*/false, /*reuse_b=*/false);
}

// A fragment: 16(M) x 32(K) f16, source row-major p[m*lda + k].
// Contiguous k order: lane half h holds k = 16h..16h+15 (matches B loaders).
static __device__ __forceinline__ v16h load_a_frag(const _Float16* __restrict__ p, int lda) {
  int lane = threadIdx.x & 31;
  int m    = lane & 15;
  int khi  = (lane >> 4) << 4;
  v16h a;
#pragma unroll
  for (int h = 0; h < 16; ++h) {
    int v = h >> 1;
    int k = h + khi; (void)v;
    a[h] = p[(size_t)m * lda + k];
  }
  return a;
}

// B fragment: 32(K) x 16(N) f16, source element (k,n) at p[k*ldb + n].
static __device__ __forceinline__ v16h load_b_frag(const _Float16* __restrict__ p, int ldb) {
  int lane = threadIdx.x & 31;
  int n    = lane & 15;
  int khi  = (lane >> 4) << 4;
  v16h b;
#pragma unroll
  for (int h = 0; h < 16; ++h)
    b[h] = p[(size_t)(h + khi) * ldb + n];
  return b;
}

// B fragment from transposed source: element (k,n) at p[n*ld + k]
// (source stored row-major as (N,K) -- used for X @ W^T GEMMs).
static __device__ __forceinline__ v16h load_bT_frag(const _Float16* __restrict__ p, int ld) {
  int lane = threadIdx.x & 31;
  int n    = lane & 15;
  int khi  = (lane >> 4) << 4;
  v16h b;
#pragma unroll
  for (int h = 0; h < 16; ++h)
    b[h] = p[(size_t)n * ld + (h + khi)];
  return b;
}

// ---------------------------------------------------------------------------
// fp32 -> fp16 conversion (grid-stride)
// ---------------------------------------------------------------------------
__global__ __launch_bounds__(256) void cvt_f32_f16(const float* __restrict__ src,
                                                   _Float16* __restrict__ dst, int n) {
  for (int i = blockIdx.x * blockDim.x + threadIdx.x; i < n; i += gridDim.x * blockDim.x)
    dst[i] = (_Float16)src[i];
}

// ---------------------------------------------------------------------------
// C = A(MxK,f16) * B(NxK,f16)^T, f32 out. One wave per 16(M) x 64(N) tile:
// one A fragment feeds 4 WMMAs per k-step (4x A reuse). Operands L2-resident.
// ---------------------------------------------------------------------------
__global__ __launch_bounds__(256) void wmma_gemm_nt(const _Float16* __restrict__ A,
                                                    const _Float16* __restrict__ Bw,
                                                    float* __restrict__ C,
                                                    int M, int N, int K) {
  int wid   = blockIdx.x * (blockDim.x >> 5) + (threadIdx.x >> 5);
  int ntile = N >> 6;
  int mt = wid / ntile, nt = wid % ntile;
  if (mt >= (M >> 4)) return;
  const _Float16* Ap = A + (size_t)(mt * 16) * K;
  v8f acc[4] = {};
  for (int k = 0; k < K; k += 32) {
    v16h a = load_a_frag(Ap + k, K);
#pragma unroll
    for (int j = 0; j < 4; ++j) {
      v16h b = load_bT_frag(Bw + (size_t)(nt * 64 + j * 16) * K + k, K);
      acc[j] = wmma_f16(a, b, acc[j]);
    }
  }
  int lane = threadIdx.x & 31;
  int n = lane & 15, mb = (lane >> 4) << 3;
#pragma unroll
  for (int j = 0; j < 4; ++j)
#pragma unroll
    for (int e = 0; e < 8; ++e)
      C[(size_t)(mt * 16 + mb + e) * N + nt * 64 + j * 16 + n] = acc[j][e];
}

// ---------------------------------------------------------------------------
// out = omega + force @ W_out^T + b_out   (f32 output to d_out)
// ---------------------------------------------------------------------------
__global__ __launch_bounds__(256) void wmma_gemm_out(const _Float16* __restrict__ A,
                                                     const _Float16* __restrict__ Bw,
                                                     const float* __restrict__ omega,
                                                     const float* __restrict__ bias,
                                                     float* __restrict__ C,
                                                     int M, int N, int K) {
  int wid   = blockIdx.x * (blockDim.x >> 5) + (threadIdx.x >> 5);
  int ntile = N >> 6;
  int mt = wid / ntile, nt = wid % ntile;
  if (mt >= (M >> 4)) return;
  const _Float16* Ap = A + (size_t)(mt * 16) * K;
  v8f acc[4] = {};
  for (int k = 0; k < K; k += 32) {
    v16h a = load_a_frag(Ap + k, K);
#pragma unroll
    for (int j = 0; j < 4; ++j) {
      v16h b = load_bT_frag(Bw + (size_t)(nt * 64 + j * 16) * K + k, K);
      acc[j] = wmma_f16(a, b, acc[j]);
    }
  }
  int lane = threadIdx.x & 31;
  int n = lane & 15, mb = (lane >> 4) << 3;
#pragma unroll
  for (int j = 0; j < 4; ++j) {
    int col  = nt * 64 + j * 16 + n;
    float add = omega[col] + bias[col];
#pragma unroll
    for (int e = 0; e < 8; ++e)
      C[(size_t)(mt * 16 + mb + e) * N + col] = acc[j][e] + add;
  }
}

// ---------------------------------------------------------------------------
// Split qk (B*T x 1024, f32) into q/k, L2-normalize per 64-wide head chunk,
// write f16 in head-major (b,h,t,d) layout. One wave per row of 64.
// ---------------------------------------------------------------------------
__global__ __launch_bounds__(256) void l2norm_split(const float* __restrict__ qk,
                                                    _Float16* __restrict__ qh,
                                                    _Float16* __restrict__ kh) {
  const int rows = BB * TT * HH;
  int wid  = blockIdx.x * (blockDim.x >> 5) + (threadIdx.x >> 5);
  int lane = threadIdx.x & 31;
  int which = (wid >= rows) ? 1 : 0;          // 0 = q, 1 = k
  int r  = which ? wid - rows : wid;
  int bt = r >> 3;                             // b*T + t
  int h  = r & 7;
  const float* src = qk + (size_t)bt * (2 * DD) + which * DD + h * HD;
  float x0 = src[lane], x1 = src[lane + 32];
  float ss = x0 * x0 + x1 * x1;
#pragma unroll
  for (int m = 16; m >= 1; m >>= 1) ss += __shfl_xor(ss, m, 32);
  float inv = 1.0f / fmaxf(sqrtf(ss), 1e-12f);
  int b = bt >> 11, t = bt & (TT - 1);
  _Float16* dst = (which ? kh : qh) + ((size_t)(b * HH + h) * TT + t) * HD;
  dst[lane]      = (_Float16)(x0 * inv);
  dst[lane + 32] = (_Float16)(x1 * inv);
}

// ---------------------------------------------------------------------------
// Fused flash-style: force[b,t,h,:] = sum_key sin(q[t]·k[key]) * k[key].
// Block = 8 waves x 16 q-rows = 128 q-rows per (b,h). K-tiles double-buffered
// in LDS via GLOBAL_LOAD_ASYNC_TO_LDS_B128 (ASYNCcnt-gated) when available.
// sin() applied in the f32 accumulator; C-layout -> A-layout conversion of
// the score tile goes through a per-wave LDS scratch (in-order DS ops).
// ---------------------------------------------------------------------------
__global__ __launch_bounds__(256) void flash_sin(const _Float16* __restrict__ qh,
                                                 const _Float16* __restrict__ kh,
                                                 _Float16* __restrict__ force) {
  __shared__ _Float16 ldsK[2][32 * HD];        // double-buffered key tile (8 KB)
  __shared__ _Float16 ldsS[8][16 * 32];        // per-wave sin(S) scratch (8 KB)

  int bh   = blockIdx.x;                       // b*H + h
  int w    = threadIdx.x >> 5;
  int lane = threadIdx.x & 31;
  int qrow = blockIdx.y * 128 + w * 16;

  const _Float16* qp = qh + (size_t)bh * TT * HD;
  const _Float16* kp = kh + (size_t)bh * TT * HD;

  // Q fragments for this wave's 16 rows, held in registers the whole loop.
  v16h aq0 = load_a_frag(qp + (size_t)qrow * HD,      HD);   // d = 0..31
  v16h aq1 = load_a_frag(qp + (size_t)qrow * HD + 32, HD);   // d = 32..63

  v8f acc0 = {}, acc1 = {}, acc2 = {}, acc3 = {};

  const int nb = TT / 32;                      // 64 key tiles
#if ASYNC_LDS
  // prefetch tile 0: 256 threads x 16B = 4 KB tile
  async_copy_b128(&ldsK[0][threadIdx.x * 8], kp + threadIdx.x * 8);
#endif

  for (int i = 0; i < nb; ++i) {
    const _Float16* kt = &ldsK[i & 1][0];
#if ASYNC_LDS
    if (i + 1 < nb) {
      async_copy_b128(&ldsK[(i + 1) & 1][threadIdx.x * 8],
                      kp + (size_t)(i + 1) * 32 * HD + threadIdx.x * 8);
      wait_async_le<1>();                      // oldest (current tile) done
    } else {
      wait_async_le<0>();
    }
    __syncthreads();                           // all waves' DMA done
#else
    __syncthreads();
    ((float4*)kt)[threadIdx.x] = ((const float4*)(kp + (size_t)i * 32 * HD))[threadIdx.x];
    __syncthreads();
#endif

    int n = lane & 15, mb = (lane >> 4) << 3;
#pragma unroll
    for (int j = 0; j < 2; ++j) {              // two 16-key subtiles
      v8f s = {};
      v16h b0 = load_bT_frag(kt + j * 16 * HD,      HD);     // B[d][key], d 0..31
      s = wmma_f16(aq0, b0, s);
      v16h b1 = load_bT_frag(kt + j * 16 * HD + 32, HD);     // d 32..63
      s = wmma_f16(aq1, b1, s);
#pragma unroll
      for (int e = 0; e < 8; ++e)              // sin + f16, store in A-layout source
        ldsS[w][(mb + e) * 32 + j * 16 + n] = (_Float16)__sinf(s[e]);
    }
    asm volatile("" ::: "memory");             // order DS store -> DS load (same wave, in-order)

    v16h as = load_a_frag(&ldsS[w][0], 32);    // sin(S): 16 q x 32 keys
    acc0 = wmma_f16(as, load_b_frag(kt +  0, HD), acc0);
    acc1 = wmma_f16(as, load_b_frag(kt + 16, HD), acc1);
    acc2 = wmma_f16(as, load_b_frag(kt + 32, HD), acc2);
    acc3 = wmma_f16(as, load_b_frag(kt + 48, HD), acc3);
    __syncthreads();                           // reads done before buffer reuse
  }

  int n = lane & 15, mb = (lane >> 4) << 3;
  int b = bh >> 3, h = bh & 7;
#pragma unroll
  for (int e = 0; e < 8; ++e) {
    int t = qrow + mb + e;
    _Float16* dst = force + ((size_t)(b * TT + t) * HH + h) * HD;  // (b,t,h,d)
    dst[n]      = (_Float16)acc0[e];
    dst[16 + n] = (_Float16)acc1[e];
    dst[32 + n] = (_Float16)acc2[e];
    dst[48 + n] = (_Float16)acc3[e];
  }
}

// ---------------------------------------------------------------------------
extern "C" void kernel_launch(void* const* d_in, const int* in_sizes, int n_in,
                              void* d_out, int out_size, void* d_ws, size_t ws_size,
                              hipStream_t stream) {
  (void)in_sizes; (void)n_in; (void)out_size; (void)ws_size;
  const float* z     = (const float*)d_in[1];   // (B,T,D)
  const float* omega = (const float*)d_in[2];   // (1,1,D)
  const float* Wqk   = (const float*)d_in[3];   // (2D,D)
  const float* Wout  = (const float*)d_in[4];   // (D,D)
  const float* bout  = (const float*)d_in[5];   // (D,)
  float* out = (float*)d_out;

  char* ws = (char*)d_ws;
  size_t off = 0;
  auto carve = [&](size_t bytes) -> void* {
    void* p = ws + off;
    off = (off + bytes + 255) & ~(size_t)255;
    return p;
  };
  const size_t NZ   = (size_t)BB * TT * DD;     // 4,194,304
  const size_t NWQK = (size_t)2 * DD * DD;      // 524,288
  const size_t NWO  = (size_t)DD * DD;          // 262,144
  _Float16* zf    = (_Float16*)carve(NZ * 2);
  _Float16* wqkf  = (_Float16*)carve(NWQK * 2);
  _Float16* woutf = (_Float16*)carve(NWO * 2);
  float*    qkf   = (float*)   carve((size_t)BB * TT * 2 * DD * 4);  // 32 MB
  _Float16* qf    = (_Float16*)carve(NZ * 2);
  _Float16* kf    = (_Float16*)carve(NZ * 2);
  _Float16* ff    = (_Float16*)carve(NZ * 2);

  // 1) fp16 conversions
  cvt_f32_f16<<<1024, 256, 0, stream>>>(z,    zf,    (int)NZ);
  cvt_f32_f16<<<512,  256, 0, stream>>>(Wqk,  wqkf,  (int)NWQK);
  cvt_f32_f16<<<256,  256, 0, stream>>>(Wout, woutf, (int)NWO);

  // 2) qk = z @ W_qk^T : M=8192, N=1024, K=512 -> 512 x 16 wave-tiles / 8
  wmma_gemm_nt<<<1024, 256, 0, stream>>>(zf, wqkf, qkf, BB * TT, 2 * DD, DD);

  // 3) split + l2-normalize into head-major q/k (131072 rows / 8 waves)
  l2norm_split<<<16384, 256, 0, stream>>>(qkf, qf, kf);

  // 4) fused sin-attention: grid (B*H, T/128)
  dim3 fgrid(BB * HH, TT / 128);
  flash_sin<<<fgrid, 256, 0, stream>>>(qf, kf, ff);

  // 5) out = omega + force @ W_out^T + b_out : 512 x 8 wave-tiles / 8
  wmma_gemm_out<<<512, 256, 0, stream>>>(ff, woutf, omega, bout, out,
                                         BB * TT, DD, DD);
}
